// SelfAttnPropagation_5265629905492
// MI455X (gfx1250) — matrix operand order repaired and checked by reading, not compiled, not run on hardware
//
#include <hip/hip_runtime.h>
#include <hip/hip_bf16.h>

// SelfAttnPropagation for MI455X (gfx1250, wave32, WMMA, async-to-LDS).
// Pipeline: transpose+f16 -> Wq/Wk f16 -> Q=proj(X) -> K=proj(Q) -> flash attn.

typedef __attribute__((ext_vector_type(16))) _Float16 v16h;
typedef __attribute__((ext_vector_type(8)))  _Float16 v8h;
typedef __attribute__((ext_vector_type(8)))  float    v8f;
typedef __attribute__((ext_vector_type(4)))  int      v4i;

#define BATCH 4
#define CH    128
#define NPIX  4096          // 64*64
#define BN    (BATCH*NPIX)  // 16384 rows total
#define KBLK  64            // keys staged per LDS block
#define ROWB  272           // padded K-row pitch in LDS (bytes): 256 + 16 pad

#define AS1 __attribute__((address_space(1)))
#define AS3 __attribute__((address_space(3)))

#if defined(__gfx1250__) && __has_builtin(__builtin_amdgcn_global_load_async_to_lds_b128)
#define USE_ASYNC_LDS 1
#else
#define USE_ASYNC_LDS 0
#endif

union V16U { v16h v; v8h h[2]; };

// ---------------------------------------------------------------------------
// Kernel 1: feature0 [b, c, n] f32  ->  Xh [b*n, c] f16  (coalesced via LDS)
// ---------------------------------------------------------------------------
__global__ void transpose_to_f16(const float* __restrict__ src, _Float16* __restrict__ dst) {
    __shared__ float tile[32][33];
    const int b  = blockIdx.z;
    const int n0 = blockIdx.x * 32;
    const int c0 = blockIdx.y * 32;
    const int tx = threadIdx.x;   // 0..31
    const int ty = threadIdx.y;   // 0..7
#pragma unroll
    for (int i = 0; i < 4; ++i) {
        int cc = c0 + ty + 8 * i;
        tile[ty + 8 * i][tx] = src[((size_t)(b * CH + cc)) * NPIX + n0 + tx];
    }
    __syncthreads();
#pragma unroll
    for (int i = 0; i < 4; ++i) {
        int nn = n0 + ty + 8 * i;
        dst[((size_t)(b * NPIX + nn)) * CH + c0 + tx] = (_Float16)tile[tx][ty + 8 * i];
    }
}

// ---------------------------------------------------------------------------
// Kernel 1b: elementwise f32 -> f16 (for the 128x128 weight matrices)
// ---------------------------------------------------------------------------
__global__ void cvt_f32_to_f16(const float* __restrict__ src, _Float16* __restrict__ dst) {
    const int i = blockIdx.x * blockDim.x + threadIdx.x;
    dst[i] = (_Float16)src[i];
}

// ---------------------------------------------------------------------------
// Kernel 2: Out[m, o] = sum_k In[m, k] * Wh[o, k] + bias[o]   (f16 in/out, f32 acc)
// One wave per 16-row tile; 8 column tiles; K=128 = 4 x wmma(16x16x32).
// ---------------------------------------------------------------------------
__global__ void proj_gemm(const _Float16* __restrict__ In, const _Float16* __restrict__ Wh,
                          const float* __restrict__ bias, _Float16* __restrict__ Out) {
    const int tid  = blockIdx.x * blockDim.x + threadIdx.x;
    const int wid  = tid >> 5;
    const int lane = threadIdx.x & 31;
    const int half = lane >> 4;
    const int l16  = lane & 15;
    const int rowBase = wid * 16;

    // A fragments: row M = l16 ; half0 K={0..15}, half1 K={16..31} (+kk*32)
    v16h af[4];
    const _Float16* arow = In + (size_t)(rowBase + l16) * CH;
#pragma unroll
    for (int kk = 0; kk < 4; ++kk) {
        const int k0 = kk * 32 + (half ? 16 : 0);
        V16U t;
        t.h[0] = *(const v8h*)(arow + k0);
        t.h[1] = *(const v8h*)(arow + k0 + 8);
        af[kk] = t.v;
    }

#pragma unroll
    for (int nb = 0; nb < 8; ++nb) {
        v8f acc = {};
#pragma unroll
        for (int kk = 0; kk < 4; ++kk) {
            // B fragment: column N = l16 -> row (nb*16+l16) of Wh; contiguous K slice
            const v16h bf = *(const v16h*)(Wh + (size_t)(nb * 16 + l16) * CH
                                              + kk * 32 + (half ? 16 : 0));
            acc = __builtin_amdgcn_wmma_f32_16x16x32_f16(false, af[kk], false, bf,
                                                         (short)0, acc, false, false);
        }
        const int col = nb * 16 + l16;
        const float bv = bias[col];
#pragma unroll
        for (int r = 0; r < 8; ++r) {
            Out[(size_t)(rowBase + r + 8 * half) * CH + col] = (_Float16)(acc[r] + bv);
        }
    }
}

// ---------------------------------------------------------------------------
// K-block staging: 64 rows x 256 B into LDS (row pitch 272 B, bank-conflict free)
// ---------------------------------------------------------------------------
__device__ inline void stage_kblock(const _Float16* __restrict__ KbBatch, int J,
                                    _Float16* lbuf, int tid) {
    const char* gb = (const char*)KbBatch + (size_t)J * (CH * 2);
    char* lb = (char*)lbuf;
#pragma unroll
    for (int i = 0; i < 8; ++i) {
        const int c   = i * 128 + tid;   // 0..1023 16-byte chunks
        const int row = c >> 4;
        const int o16 = (c & 15) * 16;
#if USE_ASYNC_LDS
        __builtin_amdgcn_global_load_async_to_lds_b128(
            (AS1 v4i*)(gb + row * 256 + o16),
            (AS3 v4i*)(lb + row * ROWB + o16),
            0, 0);
#else
        *(float4*)(lb + row * ROWB + o16) = *(const float4*)(gb + row * 256 + o16);
#endif
    }
}

__device__ inline void finish_stage() {
#if USE_ASYNC_LDS
#if __has_builtin(__builtin_amdgcn_s_wait_asynccnt)
    __builtin_amdgcn_s_wait_asynccnt(0);
#else
    asm volatile("s_wait_asynccnt 0x0" ::: "memory");
#endif
#endif
}

// ---------------------------------------------------------------------------
// Kernel 3: flash attention. 4 waves/block share double-buffered K blocks in LDS.
// One wave per 16-row query tile; online softmax; V has 2 channels.
// ---------------------------------------------------------------------------
__global__ __launch_bounds__(128) void
flash_attn(const _Float16* __restrict__ Qh, const _Float16* __restrict__ Kh,
           const float* __restrict__ flow, float* __restrict__ out) {
    __shared__ _Float16 kbuf[2][KBLK * (ROWB / 2)];

    const int tidB = threadIdx.x;                       // 0..127
    const int tid  = blockIdx.x * blockDim.x + tidB;
    const int wid  = tid >> 5;
    const int lane = tidB & 31;
    const int half = lane >> 4;
    const int l16  = lane & 15;

    const int b     = wid >> 8;        // 256 query tiles per batch; 4 waves/block same b
    const int qBase = wid * 16;        // global row base (contiguous over batches)

    const _Float16* KbBatch = Kh + (size_t)b * NPIX * CH;
    const float* f0 = flow + (size_t)(b * 2 + 0) * NPIX;
    const float* f1 = flow + (size_t)(b * 2 + 1) * NPIX;

    // Q fragments (resident for whole kernel)
    v16h qf[4];
    const _Float16* qrow = Qh + (size_t)(qBase + l16) * CH;
#pragma unroll
    for (int kk = 0; kk < 4; ++kk) {
        const int k0 = kk * 32 + (half ? 8 : 0);
        V16U t;
        t.h[0] = *(const v8h*)(qrow + k0);
        t.h[1] = *(const v8h*)(qrow + k0 + 16);
        qf[kk] = t.v;
    }

    // Online-softmax state: m replicated per half-wave; l/a0/a1 distributed per lane
    float m[8], l[8], a0[8], a1[8];
#pragma unroll
    for (int r = 0; r < 8; ++r) { m[r] = -3.0e38f; l[r] = 0.0f; a0[r] = 0.0f; a1[r] = 0.0f; }

    const float scale = 0.08838834764831845f;  // 1/sqrt(128)

    // Prologue: stage block 0
    int cur = 0;
    stage_kblock(KbBatch, 0, kbuf[cur], tidB);
    finish_stage();
    __syncthreads();

    for (int J = 0; J < NPIX; J += KBLK) {
        const int Jn = J + KBLK;
        if (Jn < NPIX) stage_kblock(KbBatch, Jn, kbuf[cur ^ 1], tidB);  // overlaps compute

        float s[4][8];
        float v0[4], v1[4];
        const char* lbase = (const char*)kbuf[cur];
#pragma unroll
        for (int t = 0; t < 4; ++t) {
            const char* lrow = lbase + (size_t)(t * 16 + l16) * ROWB + (half ? 16 : 0);
            v8f acc = {};
#pragma unroll
            for (int kk = 0; kk < 4; ++kk) {
                V16U bt;
                bt.h[0] = *(const v8h*)(lrow + kk * 64);
                bt.h[1] = *(const v8h*)(lrow + kk * 64 + 32);
                acc = __builtin_amdgcn_wmma_f32_16x16x32_f16(false, qf[kk], false, bt.v,
                                                             (short)0, acc, false, false);
            }
#pragma unroll
            for (int r = 0; r < 8; ++r) s[t][r] = acc[r] * scale;
            const int j = J + t * 16 + l16;
            v0[t] = f0[j];
            v1[t] = f1[j];
        }

#pragma unroll
        for (int r = 0; r < 8; ++r) {
            // block max over 64 keys: 4-way in-lane, then 16-lane xor reduce (stays in half)
            float tm = fmaxf(fmaxf(s[0][r], s[1][r]), fmaxf(s[2][r], s[3][r]));
            tm = fmaxf(tm, __shfl_xor(tm, 1, 32));
            tm = fmaxf(tm, __shfl_xor(tm, 2, 32));
            tm = fmaxf(tm, __shfl_xor(tm, 4, 32));
            tm = fmaxf(tm, __shfl_xor(tm, 8, 32));
            const float mn    = fmaxf(m[r], tm);
            const float alpha = __expf(m[r] - mn);
            l[r]  *= alpha;
            a0[r] *= alpha;
            a1[r] *= alpha;
#pragma unroll
            for (int t = 0; t < 4; ++t) {
                const float p = __expf(s[t][r] - mn);
                l[r]  += p;
                a0[r] += p * v0[t];
                a1[r] += p * v1[t];
            }
            m[r] = mn;
        }

        if (Jn < NPIX) {
            finish_stage();     // my async loads for the next block are done
            __syncthreads();    // everyone's loads done + everyone done reading kbuf[cur]
        }
        cur ^= 1;
    }

    // Final cross-lane reduction of distributed partials, then write
#pragma unroll
    for (int r = 0; r < 8; ++r) {
        float ls = l[r], x0 = a0[r], x1 = a1[r];
#pragma unroll
        for (int msk = 1; msk <= 8; msk <<= 1) {
            ls += __shfl_xor(ls, msk, 32);
            x0 += __shfl_xor(x0, msk, 32);
            x1 += __shfl_xor(x1, msk, 32);
        }
        if (l16 == 0) {
            const int rowInB = (qBase - b * NPIX) + r + 8 * half;
            const float inv = 1.0f / ls;
            out[(size_t)(b * 2 + 0) * NPIX + rowInB] = x0 * inv;
            out[(size_t)(b * 2 + 1) * NPIX + rowInB] = x1 * inv;
        }
    }
}

// ---------------------------------------------------------------------------
extern "C" void kernel_launch(void* const* d_in, const int* in_sizes, int n_in,
                              void* d_out, int out_size, void* d_ws, size_t ws_size,
                              hipStream_t stream) {
    const float* feature0 = (const float*)d_in[0];
    const float* flow     = (const float*)d_in[1];
    const float* Wq       = (const float*)d_in[2];
    const float* bq       = (const float*)d_in[3];
    const float* Wk       = (const float*)d_in[4];
    const float* bk       = (const float*)d_in[5];
    float* out            = (float*)d_out;

    _Float16* buf0 = (_Float16*)d_ws;                 // Xh, later reused for Kh
    _Float16* buf1 = buf0 + (size_t)BN * CH;          // Qh
    _Float16* Whq  = buf1 + (size_t)BN * CH;          // Wq in f16 (32 KB)
    _Float16* Whk  = Whq + (size_t)CH * CH;           // Wk in f16 (32 KB)

    // 1) transpose + convert: feature0 [b,c,n] -> Xh [bn, c] f16
    transpose_to_f16<<<dim3(NPIX / 32, CH / 32, BATCH), dim3(32, 8), 0, stream>>>(feature0, buf0);

    // 1b) weights f32 -> f16 once
    cvt_f32_to_f16<<<dim3((CH * CH) / 256), dim3(256), 0, stream>>>(Wq, Whq);
    cvt_f32_to_f16<<<dim3((CH * CH) / 256), dim3(256), 0, stream>>>(Wk, Whk);

    // 2) Q = X * Wq^T + bq   (1024 waves -> 256 blocks x 4 waves)
    proj_gemm<<<dim3(BN / 16 / 4), dim3(128), 0, stream>>>(buf0, Whq, bq, buf1);

    // 3) K = Q * Wk^T + bk   (reference applies k_proj to the query)
    proj_gemm<<<dim3(BN / 16 / 4), dim3(128), 0, stream>>>(buf1, Whk, bk, buf0);

    // 4) out = softmax(Q K^T / sqrt(c)) * V,  V = flow [b,2,n]
    flash_attn<<<dim3(BN / 16 / 4), dim3(128), 0, stream>>>(buf1, buf0, flow, out);
}
